// MultiHeadAttention_20564303413274
// MI455X (gfx1250) — matrix operand order repaired and checked by reading, not compiled, not run on hardware
//
#include <hip/hip_runtime.h>
#include <stdint.h>

// ---------------------------------------------------------------------------
// Types / fragment helpers for V_WMMA_F32_16X16X32_BF16 (wave32)
// ---------------------------------------------------------------------------
typedef __attribute__((ext_vector_type(16))) __bf16 bf16x16;
typedef __attribute__((ext_vector_type(8)))  float  f32x8;

union FragAB {
    bf16x16 v;
    uint32_t u[8];
    uint4    q[2];
};

__device__ __forceinline__ uint16_t f2bf(float f) {
    union { float f; uint32_t u; } c; c.f = f;
    uint32_t u = c.u;
    uint32_t r = u + 0x7FFFu + ((u >> 16) & 1u);   // round-to-nearest-even
    return (uint16_t)(r >> 16);
}

// A-matrix 16x32 bf16: lane = 16*g + m, half g holds K in [8g,8g+8) and [16+8g,16+8g+8)
__device__ __forceinline__ FragAB load_afrag(const uint16_t* rowBase, int g) {
    FragAB f;
    f.q[0] = *(const uint4*)(rowBase + 8 * g);
    f.q[1] = *(const uint4*)(rowBase + 16 + 8 * g);
    return f;
}
// B-matrix 32x16 bf16 staged as [n][k] rows: lane = 16*g + n, half g holds K in [8g,8g+8) and [16+8g,16+8g+8)
__device__ __forceinline__ FragAB load_bfrag(const uint16_t* rowBase, int g) {
    FragAB f;
    f.q[0] = *(const uint4*)(rowBase + 8 * g);
    f.q[1] = *(const uint4*)(rowBase + 16 + 8 * g);
    return f;
}

// ---------------------------------------------------------------------------
// Async global->LDS copy of an 8KB tile (rows of ROW_ELEMS bf16), 256 threads,
// 2 x GLOBAL_LOAD_ASYNC_TO_LDS_B128 per thread (tracked by ASYNCcnt).
// Flat LDS pointers carry the LDS byte offset in addr[31:0].
// ---------------------------------------------------------------------------
template <int ROW_ELEMS>
__device__ __forceinline__ void async_tile_8k(uint16_t* lds_tile,
                                              const uint16_t* g_base,
                                              size_t g_row_stride, int tid) {
    constexpr int CPR = ROW_ELEMS / 8;   // 16-byte chunks per row
#pragma unroll
    for (int it = 0; it < 2; ++it) {
        int cid = tid + it * 256;
        int r = cid / CPR;
        int c = (cid % CPR) * 8;
        uint32_t lo = (uint32_t)(uintptr_t)(lds_tile + r * ROW_ELEMS + c);
        uint64_t ga = (uint64_t)(uintptr_t)(g_base + (size_t)r * g_row_stride + c);
        asm volatile("global_load_async_to_lds_b128 %0, %1, off"
                     :: "v"(lo), "v"(ga) : "memory");
    }
}
__device__ __forceinline__ void wait_async0() {
    asm volatile("s_wait_asynccnt 0x0" ::: "memory");
}

// Load 16 contiguous bf16 (32B) into regs; scatter transposed [n][k] into LDS.
__device__ __forceinline__ void load_b_regs(const uint16_t* src, uint4& r0, uint4& r1) {
    r0 = *(const uint4*)src;
    r1 = *(const uint4*)(src + 8);
}
__device__ __forceinline__ void scatter_t(uint16_t* lds, int kr, int n0,
                                          const uint4& r0, const uint4& r1) {
    const uint16_t* p0 = (const uint16_t*)&r0;
    const uint16_t* p1 = (const uint16_t*)&r1;
#pragma unroll
    for (int e = 0; e < 8; ++e) {
        lds[(n0 + e) * 32 + kr]     = p0[e];
        lds[(n0 + 8 + e) * 32 + kr] = p1[e];
    }
}

// ---------------------------------------------------------------------------
// fp32 -> bf16 conversion (vectorized x4)
// ---------------------------------------------------------------------------
__global__ void cvt_f32_to_bf16(const float* __restrict__ in,
                                uint16_t* __restrict__ out, int n4) {
    int i = blockIdx.x * blockDim.x + threadIdx.x;
    int stride = gridDim.x * blockDim.x;
    const float4* in4 = (const float4*)in;
    for (; i < n4; i += stride) {
        float4 v = in4[i];
        union { uint16_t h[4]; uint2 u2; } o;
        o.h[0] = f2bf(v.x); o.h[1] = f2bf(v.y);
        o.h[2] = f2bf(v.z); o.h[3] = f2bf(v.w);
        *(uint2*)(out + (size_t)i * 4) = o.u2;
    }
}

// ---------------------------------------------------------------------------
// Tiled GEMM: C[M,N] = A[M,K] * B[K,N], A/B bf16, accumulate f32.
// Block tile 128x128, K-step 32, 256 threads = 8 waves, wave tile 32x64.
// Double-buffered LDS; A tile via async DMA, B tile via reg-staged transpose.
// ---------------------------------------------------------------------------
template <bool OUT_BF16>
__global__ __launch_bounds__(256)
void gemm_bf16_wmma(const uint16_t* __restrict__ A,
                    const uint16_t* __restrict__ Bm,
                    void* __restrict__ Cout, int M, int N, int K) {
    __shared__ __attribute__((aligned(16))) uint16_t lA[2][128 * 32]; // [m][k]
    __shared__ __attribute__((aligned(16))) uint16_t lB[2][128 * 32]; // [n][k]

    const int tid  = threadIdx.x;
    const int lane = tid & 31;
    const int wv   = tid >> 5;
    const int g    = (lane >> 4) & 1;
    const int ln   = lane & 15;
    const int mBlock = blockIdx.y * 128;
    const int nBlock = blockIdx.x * 128;
    const int wm = wv & 3;   // 4 wave-rows of 32
    const int wn = wv >> 2;  // 2 wave-cols of 64
    const int kr = tid >> 3;        // 0..31 : B k-row
    const int n0 = (tid & 7) * 16;  // B n-group

    f32x8 acc[2][4] = {};
    uint4 br0, br1;

    // prologue: stage tile 0 into buffer 0
    async_tile_8k<32>(lA[0], A + (size_t)mBlock * K, K, tid);
    load_b_regs(Bm + (size_t)kr * N + nBlock + n0, br0, br1);
    scatter_t(lB[0], kr, n0, br0, br1);
    wait_async0();
    __syncthreads();

    int buf = 0;
    for (int k0 = 0; k0 < K; k0 += 32) {
        const bool hasNext = (k0 + 32) < K;
        if (hasNext) {   // issue next tile's traffic before computing
            async_tile_8k<32>(lA[buf ^ 1], A + (size_t)mBlock * K + k0 + 32, K, tid);
            load_b_regs(Bm + (size_t)(k0 + 32 + kr) * N + nBlock + n0, br0, br1);
        }

        FragAB afr[2], bfr[4];
#pragma unroll
        for (int i = 0; i < 2; ++i)
            afr[i] = load_afrag(&lA[buf][(wm * 32 + i * 16 + ln) * 32], g);
#pragma unroll
        for (int j = 0; j < 4; ++j)
            bfr[j] = load_bfrag(&lB[buf][(wn * 64 + j * 16 + ln) * 32], g);
#pragma unroll
        for (int i = 0; i < 2; ++i)
#pragma unroll
            for (int j = 0; j < 4; ++j)
                acc[i][j] = __builtin_amdgcn_wmma_f32_16x16x32_bf16(
                    false, afr[i].v, false, bfr[j].v, (short)0, acc[i][j],
                    false, false);

        if (hasNext) scatter_t(lB[buf ^ 1], kr, n0, br0, br1);
        wait_async0();
        __syncthreads();
        buf ^= 1;
    }

    // C/D layout: lane 16g+n holds rows (8g + r), col n, r = vgpr index
#pragma unroll
    for (int i = 0; i < 2; ++i) {
        int mb = mBlock + wm * 32 + i * 16 + g * 8;
#pragma unroll
        for (int j = 0; j < 4; ++j) {
            int nb = nBlock + wn * 64 + j * 16 + ln;
#pragma unroll
            for (int r = 0; r < 8; ++r) {
                float val = acc[i][j][r];
                if (OUT_BF16)
                    ((uint16_t*)Cout)[(size_t)(mb + r) * N + nb] = f2bf(val);
                else
                    ((float*)Cout)[(size_t)(mb + r) * N + nb] = val;
            }
        }
    }
}

// ---------------------------------------------------------------------------
// Flash attention with the reference's multiplicative tril mask and 1/sqrt(T)
// scale. Q,K,V bf16 [B*T, D] (head = contiguous Dh slice). Out bf16 same shape.
// Grid: (T/128, H, B); 256 threads; each wave owns 16 query rows.
// Double-buffered K (async DMA) and V^T (reg-staged transpose) tiles.
// ---------------------------------------------------------------------------
__global__ __launch_bounds__(256)
void flash_attn_wmma(const uint16_t* __restrict__ Qg,
                     const uint16_t* __restrict__ Kg,
                     const uint16_t* __restrict__ Vg,
                     uint16_t* __restrict__ Og,
                     int T, int D, int H) {
    const int Dh   = D / H;            // 128
    const int h    = blockIdx.y;
    const int b    = blockIdx.z;
    const int tid  = threadIdx.x;
    const int lane = tid & 31;
    const int wv   = tid >> 5;
    const int g    = (lane >> 4) & 1;
    const int ln   = lane & 15;
    const int tBase = blockIdx.x * 128 + wv * 16;  // query rows of this wave
    const size_t baseBT = (size_t)b * T;
    const size_t rs = (size_t)D;                   // row stride in bf16

    __shared__ __attribute__((aligned(16))) uint16_t lK[2][32 * 128];   // [s][d]
    __shared__ __attribute__((aligned(16))) uint16_t lVT[2][128 * 32];  // [d][s]
    __shared__ __attribute__((aligned(16))) uint16_t lP[8][16 * 32];    // per wave

    const int sr = tid >> 3;        // 0..31 : V s-row
    const int d0 = (tid & 7) * 16;  // V d-group

    // Q fragments: 4 chunks over Dh=128, straight from global (A layout)
    FragAB qfr[4];
    {
        const uint16_t* qrow = Qg + (baseBT + tBase + ln) * rs + h * Dh;
#pragma unroll
        for (int c = 0; c < 4; ++c) qfr[c] = load_afrag(qrow + c * 32, g);
    }

    f32x8 oacc[8] = {};
    float mrow[8], lrow[8];
#pragma unroll
    for (int r = 0; r < 8; ++r) { mrow[r] = -3.0e38f; lrow[r] = 0.0f; }

    const float scale = 0.022097086912079608f;  // 1/sqrt(2048)

    // prologue: stage chunk 0 into buffer 0
    uint4 vr0, vr1;
    async_tile_8k<128>(lK[0], Kg + baseBT * rs + h * Dh, rs, tid);
    load_b_regs(Vg + (baseBT + sr) * rs + h * Dh + d0, vr0, vr1);
    scatter_t(lVT[0], sr, d0, vr0, vr1);
    wait_async0();
    __syncthreads();

    int buf = 0;
    for (int s0 = 0; s0 < T; s0 += 32) {
        const bool hasNext = (s0 + 32) < T;
        if (hasNext) {   // issue next chunk's traffic before computing
            async_tile_8k<128>(lK[buf ^ 1],
                               Kg + (baseBT + s0 + 32) * rs + h * Dh, rs, tid);
            load_b_regs(Vg + (baseBT + s0 + 32 + sr) * rs + h * Dh + d0, vr0, vr1);
        }

        // S = Q * K^T : two 16x16 f32 tiles (cols s0+16j+n)
        f32x8 s[2] = {};
#pragma unroll
        for (int j = 0; j < 2; ++j)
#pragma unroll
            for (int c = 0; c < 4; ++c) {
                FragAB kfr = load_bfrag(&lK[buf][(j * 16 + ln) * 128 + c * 32], g);
                s[j] = __builtin_amdgcn_wmma_f32_16x16x32_bf16(
                    false, qfr[c].v, false, kfr.v, (short)0, s[j], false, false);
            }

        // multiplicative tril mask (masked -> 0, still in softmax) + scale
#pragma unroll
        for (int j = 0; j < 2; ++j) {
            int scol = s0 + j * 16 + ln;
#pragma unroll
            for (int r = 0; r < 8; ++r) {
                int trow = tBase + g * 8 + r;
                float v = s[j][r] * scale;
                s[j][r] = (scol <= trow) ? v : 0.0f;
            }
        }

        // online softmax per row (rows live across 16-lane half groups)
        float alpha[8];
#pragma unroll
        for (int r = 0; r < 8; ++r) {
            float mx = fmaxf(s[0][r], s[1][r]);
#pragma unroll
            for (int off = 8; off >= 1; off >>= 1)
                mx = fmaxf(mx, __shfl_xor(mx, off, 32));
            float mnew = fmaxf(mrow[r], mx);
            alpha[r] = __expf(mrow[r] - mnew);
            mrow[r] = mnew;
            float p0 = __expf(s[0][r] - mnew);
            float p1 = __expf(s[1][r] - mnew);
            s[0][r] = p0; s[1][r] = p1;
            float ps = p0 + p1;
#pragma unroll
            for (int off = 8; off >= 1; off >>= 1)
                ps += __shfl_xor(ps, off, 32);
            lrow[r] = lrow[r] * alpha[r] + ps;
        }
#pragma unroll
        for (int dt = 0; dt < 8; ++dt)
#pragma unroll
            for (int r = 0; r < 8; ++r) oacc[dt][r] *= alpha[r];

        // P: C layout -> A layout via per-wave LDS
#pragma unroll
        for (int j = 0; j < 2; ++j)
#pragma unroll
            for (int r = 0; r < 8; ++r)
                lP[wv][(g * 8 + r) * 32 + j * 16 + ln] = f2bf(s[j][r]);
        __asm__ volatile("s_wait_dscnt 0x0" ::: "memory");

        FragAB pfr = load_afrag(&lP[wv][ln * 32], g);
#pragma unroll
        for (int dt = 0; dt < 8; ++dt) {
            FragAB vfr = load_bfrag(&lVT[buf][(dt * 16 + ln) * 32], g);
            oacc[dt] = __builtin_amdgcn_wmma_f32_16x16x32_bf16(
                false, pfr.v, false, vfr.v, (short)0, oacc[dt], false, false);
        }

        if (hasNext) scatter_t(lVT[buf ^ 1], sr, d0, vr0, vr1);
        wait_async0();
        __syncthreads();
        buf ^= 1;
    }

    // normalize and store bf16
#pragma unroll
    for (int dt = 0; dt < 8; ++dt)
#pragma unroll
        for (int r = 0; r < 8; ++r) {
            float v = oacc[dt][r] / lrow[r];
            size_t row = baseBT + tBase + g * 8 + r;
            Og[row * rs + h * Dh + dt * 16 + ln] = f2bf(v);
        }
}

// ---------------------------------------------------------------------------
// Host orchestration
// ---------------------------------------------------------------------------
extern "C" void kernel_launch(void* const* d_in, const int* in_sizes, int n_in,
                              void* d_out, int out_size, void* d_ws, size_t ws_size,
                              hipStream_t stream) {
    const int B = 2, T = 2048, D = 2048, H = 16;
    const int M = B * T;                       // 4096
    const size_t nX = (size_t)M * D;           // 8388608
    const size_t nW = (size_t)D * D;           // 4194304

    const float* x  = (const float*)d_in[0];
    const float* Wq = (const float*)d_in[1];
    const float* Wk = (const float*)d_in[2];
    const float* Wv = (const float*)d_in[3];
    const float* Wo = (const float*)d_in[4];

    char* ws = (char*)d_ws;
    uint16_t* xb  = (uint16_t*)ws;                 ws += nX * 2;
    uint16_t* Wqb = (uint16_t*)ws;                 ws += nW * 2;
    uint16_t* Wkb = (uint16_t*)ws;                 ws += nW * 2;
    uint16_t* Wvb = (uint16_t*)ws;                 ws += nW * 2;
    uint16_t* Wob = (uint16_t*)ws;                 ws += nW * 2;
    uint16_t* Qb  = (uint16_t*)ws;                 ws += nX * 2;
    uint16_t* Kb  = (uint16_t*)ws;                 ws += nX * 2;
    uint16_t* Vb  = (uint16_t*)ws;                 ws += nX * 2;
    uint16_t* Ab  = (uint16_t*)ws;                 ws += nX * 2;

    // fp32 -> bf16 (x4 vectorized)
    cvt_f32_to_bf16<<<2048, 256, 0, stream>>>(x,  xb,  (int)(nX / 4));
    cvt_f32_to_bf16<<<1024, 256, 0, stream>>>(Wq, Wqb, (int)(nW / 4));
    cvt_f32_to_bf16<<<1024, 256, 0, stream>>>(Wk, Wkb, (int)(nW / 4));
    cvt_f32_to_bf16<<<1024, 256, 0, stream>>>(Wv, Wvb, (int)(nW / 4));
    cvt_f32_to_bf16<<<1024, 256, 0, stream>>>(Wo, Wob, (int)(nW / 4));

    // Q/K/V projections (bf16 out)
    dim3 gb(D / 128, M / 128);                 // (16, 32)
    gemm_bf16_wmma<true><<<gb, 256, 0, stream>>>(xb, Wqb, Qb, M, D, D);
    gemm_bf16_wmma<true><<<gb, 256, 0, stream>>>(xb, Wkb, Kb, M, D, D);
    gemm_bf16_wmma<true><<<gb, 256, 0, stream>>>(xb, Wvb, Vb, M, D, D);

    // attention
    dim3 ga(T / 128, H, B);                    // (16, 16, 2)
    flash_attn_wmma<<<ga, 256, 0, stream>>>(Qb, Kb, Vb, Ab, T, D, H);

    // output projection (fp32 out)
    gemm_bf16_wmma<false><<<gb, 256, 0, stream>>>(Ab, Wob, (float*)d_out, M, D, D);
}
